// HPEHGTModel_89300960018747
// MI455X (gfx1250) — matrix operand order repaired and checked by reading, not compiled, not run on hardware
//
#include <hip/hip_runtime.h>
#include <hip/hip_bf16.h>
#include <math.h>

// ---------------- model constants ----------------
#define N_   4096
#define IN_  512
#define HID_ 128
#define H_   4
#define T_   3
#define E_   131072
#define SVD_ 32
#define LAP_ 16
#define L_   2
#define CLS_ 8
#define DH_  32
#define PED_ 112          // T*SVD + LAP
#define PEDP_ 128         // PED padded to K%32==0
#define H4C_ 512          // H_*HID_
#define CLSP_ 64          // CLS padded to Nc%64==0

#define GEMM_RELU 1
#define GEMM_ACC  2

typedef _Float16 half_t;
typedef __attribute__((ext_vector_type(16))) _Float16 v16h;
typedef __attribute__((ext_vector_type(8)))  float    v8f;

// ---------- ordered-uint encoding for float atomicMax ----------
__device__ __forceinline__ unsigned ordf(float f) {
    unsigned u = __float_as_uint(f);
    return (u & 0x80000000u) ? ~u : (u | 0x80000000u);
}
__device__ __forceinline__ float unordf(unsigned u) {
    return (u & 0x80000000u) ? __uint_as_float(u ^ 0x80000000u)
                             : __uint_as_float(~u);
}
#define ORD_NEG_INF 0x007FFFFFu   // ordf(-inf)

// =====================================================================
// Pack B [Ksrc,Ncsrc] (row stride ldb, f32) into f16 fragment-major
// layout for the WMMA B operand, zero-padding to [K,Nc].
// Fragment f = (kt*(Nc/16)+ct); half index within fragment:
//   Bp[(f*32 + lane)*16 + i] = B[kt*32 + (lane<16?0:8) + (i&7) + (i>=8?16:0)][ct*16 + (lane&15)]
// Matches the A fragment K order (lane=column, two 8-K runs).
// =====================================================================
__global__ void pack_b_kernel(const float* __restrict__ B, half_t* __restrict__ Bp,
                              int K, int Nc, int Ksrc, int Ncsrc, int ldb) {
    int idx = blockIdx.x * blockDim.x + threadIdx.x;
    if (idx >= K * Nc) return;
    int i    = idx & 15;
    int lane = (idx >> 4) & 31;
    int rest = idx >> 9;                 // kt*(Nc/16) + ct
    int nct  = Nc >> 4;
    int kt   = rest / nct, ct = rest - kt * nct;
    int kk   = kt * 32 + ((lane < 16) ? 0 : 8) + (i & 7) + ((i >> 3) << 4);
    int cc   = ct * 16 + (lane & 15);
    Bp[idx] = (kk < Ksrc && cc < Ncsrc) ? (half_t)B[(size_t)kk * ldb + cc]
                                        : (half_t)0.0f;
}

// =====================================================================
// WMMA GEMM: C[M,Nc] (=|+=) A[M,K] @ B[K,Nc] (+bias)(relu)
// Contract: M%16==0, K%32==0, Nc%64==0, 16B-aligned rows; B pre-packed.
// One wave computes a 16x64 tile: 1 A fragment reused by 4 WMMAs/k-step;
// each B fragment is one aligned 32-byte vector load.
// =====================================================================
__global__ void gemm_wmma_kernel(const float* __restrict__ A, int lda,
                                 const half_t* __restrict__ Bp,    // packed
                                 const float* __restrict__ bias,   // [Nc] or null
                                 float* __restrict__ C, int ldc,
                                 int K, int Nc, int flags) {
    const int lane = threadIdx.x;             // 0..31 (wave32)
    const int row0 = blockIdx.x * 16;
    const int col0 = blockIdx.y * 64;
    const int nct  = Nc >> 4;
    const int ct0  = col0 >> 4;

    const int mA  = row0 + (lane & 15);
    const int kbA = (lane < 16) ? 0 : 8;
    const int crow0 = row0 + ((lane < 16) ? 0 : 8);
    const int ccol  = lane & 15;

    v8f acc[4];
    if (flags & GEMM_ACC) {
        #pragma unroll
        for (int j = 0; j < 4; ++j)
            #pragma unroll
            for (int r = 0; r < 8; ++r)
                acc[j][r] = C[(crow0 + r) * ldc + col0 + j * 16 + ccol];
    } else {
        const v8f z = {0.f, 0.f, 0.f, 0.f, 0.f, 0.f, 0.f, 0.f};
        #pragma unroll
        for (int j = 0; j < 4; ++j) acc[j] = z;
    }

    const float* Arow = A + (size_t)mA * lda;
    for (int k0 = 0; k0 < K; k0 += 32) {
        // ---- A fragment: two contiguous 8-float runs -> 4x b128 loads ----
        const float4* p0 = (const float4*)(Arow + k0 + kbA);
        const float4* p1 = (const float4*)(Arow + k0 + 16 + kbA);
        float4 f0 = p0[0], f1 = p0[1], f2 = p1[0], f3 = p1[1];
        v16h a;
        a[0] = (half_t)f0.x; a[1] = (half_t)f0.y; a[2] = (half_t)f0.z; a[3] = (half_t)f0.w;
        a[4] = (half_t)f1.x; a[5] = (half_t)f1.y; a[6] = (half_t)f1.z; a[7] = (half_t)f1.w;
        a[8]  = (half_t)f2.x; a[9]  = (half_t)f2.y; a[10] = (half_t)f2.z; a[11] = (half_t)f2.w;
        a[12] = (half_t)f3.x; a[13] = (half_t)f3.y; a[14] = (half_t)f3.z; a[15] = (half_t)f3.w;
        // ---- 4 pre-packed B fragments, 4 WMMAs (A reused; no D hazard) ----
        const size_t fbase = ((size_t)(k0 >> 5) * nct + ct0) * 32 + lane;
        #pragma unroll
        for (int j = 0; j < 4; ++j) {
            v16h b = *(const v16h*)(Bp + ((fbase + (size_t)j * 32) << 4));
            acc[j] = __builtin_amdgcn_wmma_f32_16x16x32_f16(false, a, false, b,
                                                            (short)0, acc[j], false, false);
        }
    }

    #pragma unroll
    for (int j = 0; j < 4; ++j) {
        const int cc = col0 + j * 16 + ccol;
        const float bv = bias ? bias[cc] : 0.0f;
        #pragma unroll
        for (int r = 0; r < 8; ++r) {
            float v = acc[j][r] + bv;
            if (flags & GEMM_RELU) v = fmaxf(v, 0.0f);
            C[(crow0 + r) * ldc + cc] = v;
        }
    }
}

// =====================================================================
// Flash attention, one wave per (16 queries, 1 head).
// S tile via WMMA with keys on the M axis -> softmax over keys needs only
// in-lane max + one shfl_xor(16). P·V accumulated on VALU (DH=32).
// Q/K/V layout: [N, HID] with head h occupying columns h*32..h*32+31.
// =====================================================================
__global__ void attn_flash_kernel(const float* __restrict__ Q,
                                  const float* __restrict__ Km,
                                  const float* __restrict__ V,
                                  float* __restrict__ O,
                                  int n, float scale) {
    const int lane  = threadIdx.x;
    const int n0    = blockIdx.x * 16;
    const int h     = blockIdx.y;
    const int hb    = h * DH_;
    const int colq  = lane & 15;
    const int qrow  = n0 + colq;
    const int kbB   = (lane < 16) ? 0 : 16;   // unused; the B fragment is loaded with kbA
    const int kbA   = (lane < 16) ? 0 : 8;    // A fragment K base
    const int dbase = (lane < 16) ? 0 : 16;   // output d range for this lane

    // B = Q^T (32 x 16): two 8-float runs (same K order as A), loaded once.
    v16h bq;
    {
        const float4* qp = (const float4*)(Q + (size_t)qrow * HID_ + hb + kbA);
        const float4* qp1 = (const float4*)(Q + (size_t)qrow * HID_ + hb + 16 + kbA);
        float4 q0 = qp[0], q1 = qp[1], q2 = qp1[0], q3 = qp1[1];
        bq[0]  = (half_t)q0.x; bq[1]  = (half_t)q0.y; bq[2]  = (half_t)q0.z; bq[3]  = (half_t)q0.w;
        bq[4]  = (half_t)q1.x; bq[5]  = (half_t)q1.y; bq[6]  = (half_t)q1.z; bq[7]  = (half_t)q1.w;
        bq[8]  = (half_t)q2.x; bq[9]  = (half_t)q2.y; bq[10] = (half_t)q2.z; bq[11] = (half_t)q2.w;
        bq[12] = (half_t)q3.x; bq[13] = (half_t)q3.y; bq[14] = (half_t)q3.z; bq[15] = (half_t)q3.w;
    }

    float m_run = -INFINITY, l_run = 0.0f;
    float oacc[16];
    #pragma unroll
    for (int d = 0; d < 16; ++d) oacc[d] = 0.0f;

    for (int kb0 = 0; kb0 < n; kb0 += 16) {
        // A = 16 key rows x 32 d : two 8-float runs per lane
        const int krow = kb0 + (lane & 15);
        const float4* kp0 = (const float4*)(Km + (size_t)krow * HID_ + hb + kbA);
        const float4* kp1 = (const float4*)(Km + (size_t)krow * HID_ + hb + 16 + kbA);
        float4 f0 = kp0[0], f1 = kp0[1], f2 = kp1[0], f3 = kp1[1];
        v16h aK;
        aK[0] = (half_t)f0.x; aK[1] = (half_t)f0.y; aK[2] = (half_t)f0.z; aK[3] = (half_t)f0.w;
        aK[4] = (half_t)f1.x; aK[5] = (half_t)f1.y; aK[6] = (half_t)f1.z; aK[7] = (half_t)f1.w;
        aK[8]  = (half_t)f2.x; aK[9]  = (half_t)f2.y; aK[10] = (half_t)f2.z; aK[11] = (half_t)f2.w;
        aK[12] = (half_t)f3.x; aK[13] = (half_t)f3.y; aK[14] = (half_t)f3.z; aK[15] = (half_t)f3.w;

        v8f c = {0.f, 0.f, 0.f, 0.f, 0.f, 0.f, 0.f, 0.f};
        c = __builtin_amdgcn_wmma_f32_16x16x32_f16(false, aK, false, bq,
                                                   (short)0, c, false, false);
        // c[r] = score(key kb0 + r + (lane<16?0:8), query qrow)
        float s[8], tmax = -INFINITY;
        #pragma unroll
        for (int r = 0; r < 8; ++r) { s[r] = c[r] * scale; tmax = fmaxf(tmax, s[r]); }
        tmax = fmaxf(tmax, __shfl_xor(tmax, 16));
        const float mnew  = fmaxf(m_run, tmax);
        const float alpha = __expf(m_run - mnew);
        float p[8], psum = 0.0f;
        #pragma unroll
        for (int r = 0; r < 8; ++r) { p[r] = __expf(s[r] - mnew); psum += p[r]; }
        psum += __shfl_xor(psum, 16);
        l_run = l_run * alpha + psum;
        m_run = mnew;

        float po[8];
        #pragma unroll
        for (int r = 0; r < 8; ++r) po[r] = __shfl_xor(p[r], 16);
        float pm[16];
        if (lane < 16) {
            #pragma unroll
            for (int r = 0; r < 8; ++r) { pm[r] = p[r]; pm[8 + r] = po[r]; }
        } else {
            #pragma unroll
            for (int r = 0; r < 8; ++r) { pm[r] = po[r]; pm[8 + r] = p[r]; }
        }
        #pragma unroll
        for (int d = 0; d < 16; ++d) oacc[d] *= alpha;
        #pragma unroll
        for (int mm = 0; mm < 16; ++mm) {
            const float4* vp = (const float4*)(V + (size_t)(kb0 + mm) * HID_ + hb + dbase);
            float4 v0 = vp[0], v1 = vp[1], v2 = vp[2], v3 = vp[3];
            const float w = pm[mm];
            oacc[0]  += w * v0.x; oacc[1]  += w * v0.y; oacc[2]  += w * v0.z; oacc[3]  += w * v0.w;
            oacc[4]  += w * v1.x; oacc[5]  += w * v1.y; oacc[6]  += w * v1.z; oacc[7]  += w * v1.w;
            oacc[8]  += w * v2.x; oacc[9]  += w * v2.y; oacc[10] += w * v2.z; oacc[11] += w * v2.w;
            oacc[12] += w * v3.x; oacc[13] += w * v3.y; oacc[14] += w * v3.z; oacc[15] += w * v3.w;
        }
    }
    const float inv = 1.0f / l_run;
    #pragma unroll
    for (int d = 0; d < 16; ++d)
        O[(size_t)qrow * HID_ + hb + dbase + d] = oacc[d] * inv;
}

// =====================================================================
// GAT helpers
// =====================================================================
__global__ void gat_scores_kernel(const float* __restrict__ xw,
                                  const float* __restrict__ a_s,
                                  const float* __restrict__ a_d,
                                  float* __restrict__ ss, float* __restrict__ sd,
                                  int n, int Hh, int C) {
    int idx = blockIdx.x * blockDim.x + threadIdx.x;
    if (idx >= n * Hh) return;
    int node = idx / Hh, hh = idx - node * Hh;
    const float4* xr = (const float4*)(xw + (size_t)node * (Hh * C) + hh * C);
    const float4* ar = (const float4*)(a_s + hh * C);
    const float4* dr = (const float4*)(a_d + hh * C);
    float a = 0.0f, b = 0.0f;
    for (int c = 0; c < C / 4; ++c) {
        float4 xv = xr[c], av = ar[c], dv = dr[c];
        a += xv.x * av.x + xv.y * av.y + xv.z * av.z + xv.w * av.w;
        b += xv.x * dv.x + xv.y * dv.y + xv.z * dv.z + xv.w * dv.w;
    }
    ss[idx] = a; sd[idx] = b;
}

__global__ void gat_edge_logit_max_kernel(const long long* __restrict__ src,
                                          const long long* __restrict__ dst,
                                          const float* __restrict__ ss,
                                          const float* __restrict__ sd,
                                          float* __restrict__ exbuf,
                                          unsigned* __restrict__ emax,
                                          int E, int Hh) {
    int idx = blockIdx.x * blockDim.x + threadIdx.x;
    if (idx >= E * Hh) return;
    int e = idx / Hh, hh = idx - e * Hh;
    int s = (int)src[e], d = (int)dst[e];
    float l = ss[s * Hh + hh] + sd[d * Hh + hh];
    l = (l > 0.0f) ? l : 0.2f * l;             // leaky_relu 0.2
    exbuf[idx] = l;
    atomicMax(&emax[d * Hh + hh], ordf(l));
}

__global__ void gat_edge_exp_sum_kernel(const long long* __restrict__ dst,
                                        float* __restrict__ exbuf,
                                        const unsigned* __restrict__ emax,
                                        float* __restrict__ esum,
                                        int E, int Hh) {
    int idx = blockIdx.x * blockDim.x + threadIdx.x;
    if (idx >= E * Hh) return;
    int e = idx / Hh, hh = idx - e * Hh;
    int d = (int)dst[e];
    float ex = __expf(exbuf[idx] - unordf(emax[d * Hh + hh]));
    exbuf[idx] = ex;
    atomicAdd(&esum[d * Hh + hh], ex);
}

__global__ void gat_edge_scatter_kernel(const long long* __restrict__ src,
                                        const long long* __restrict__ dst,
                                        const float* __restrict__ xw,
                                        const float* __restrict__ exbuf,
                                        const float* __restrict__ esum,
                                        float* __restrict__ out,
                                        int Hh, int C) {
    int e = blockIdx.x;
    int s = (int)src[e], d = (int)dst[e];
    int F = Hh * C;
    for (int j = threadIdx.x; j < F; j += blockDim.x) {
        int hh = j / C;
        float a = exbuf[e * Hh + hh] / (esum[d * Hh + hh] + 1e-16f);
        atomicAdd(&out[d * F + j], a * xw[s * F + j]);
    }
}

__global__ void bias3_act_kernel(float* __restrict__ buf,
                                 const float* __restrict__ b0,
                                 const float* __restrict__ b1,
                                 const float* __restrict__ b2,
                                 int n, int F, int do_relu) {
    int idx = blockIdx.x * blockDim.x + threadIdx.x;
    if (idx >= n * F) return;
    int j = idx % F;
    float v = buf[idx] + b0[j] + b1[j] + b2[j];
    buf[idx] = do_relu ? fmaxf(v, 0.0f) : v;
}

// =====================================================================
// PE concat + reconstruction loss
// =====================================================================
__global__ void pe_concat_kernel(const float* __restrict__ peQ,
                                 const float* __restrict__ peK,
                                 const float* __restrict__ lap,
                                 float* __restrict__ pq_ws, float* __restrict__ pk_ws,
                                 int ldp,
                                 float* __restrict__ pq_out, float* __restrict__ pk_out,
                                 int n) {
    int idx = blockIdx.x * blockDim.x + threadIdx.x;
    if (idx >= n * PED_) return;
    int node = idx / PED_, j = idx - node * PED_;
    float q, k;
    if (j < T_ * SVD_) {
        int t = j / SVD_, c = j - t * SVD_;
        q = peQ[((size_t)t * n + node) * SVD_ + c];
        k = peK[((size_t)t * n + node) * SVD_ + c];
    } else {
        q = k = lap[node * LAP_ + (j - T_ * SVD_)];
    }
    pq_ws[node * ldp + j] = q; pk_ws[node * ldp + j] = k;
    pq_out[idx] = q; pk_out[idx] = k;
}

__global__ void pe_loss_kernel(const float* __restrict__ peQ,
                               const float* __restrict__ peK,
                               const float* __restrict__ A,
                               float* __restrict__ acc, int n, int total) {
    __shared__ float sh[256];
    int gid = blockIdx.x * blockDim.x + threadIdx.x;
    float sq = 0.0f;
    if (gid < total) {
        int nn = n * n;
        int t = gid / nn;
        int rem = gid - t * nn;
        int i = rem / n, j = rem - i * n;
        const float4* qr = (const float4*)(peQ + ((size_t)t * n + i) * SVD_);
        const float4* kr = (const float4*)(peK + ((size_t)t * n + j) * SVD_);
        float d = 0.0f;
        #pragma unroll
        for (int c = 0; c < SVD_ / 4; ++c) {
            float4 qv = qr[c], kv = kr[c];
            d += qv.x * kv.x + qv.y * kv.y + qv.z * kv.z + qv.w * kv.w;
        }
        float diff = d - A[gid];                // [T,N,N] flat == gid
        sq = diff * diff;
    }
    sh[threadIdx.x] = sq; __syncthreads();
    for (int s = 128; s > 0; s >>= 1) {
        if (threadIdx.x < s) sh[threadIdx.x] += sh[threadIdx.x + s];
        __syncthreads();
    }
    if (threadIdx.x == 0) atomicAdd(acc, sh[0]);
}

__global__ void loss_finalize_kernel(const float* __restrict__ acc,
                                     const float* __restrict__ alpha,
                                     float* __restrict__ out, float invscale) {
    if (threadIdx.x == 0 && blockIdx.x == 0)
        out[0] = alpha[0] * acc[0] * invscale;
}

// =====================================================================
// residual + LayerNorm (one 128-thread block per node)
// =====================================================================
__global__ void resid_ln_kernel(const float* __restrict__ hh, int ldh,
                                const float* __restrict__ o,
                                const float* __restrict__ deg,   // null -> 1.0
                                const float* __restrict__ g,
                                const float* __restrict__ b,
                                float* __restrict__ out, int ldo) {
    __shared__ float sh[HID_];
    int node = blockIdx.x, j = threadIdx.x;
    float val = hh[node * ldh + j] + o[node * HID_ + j] * (deg ? deg[node] : 1.0f);
    sh[j] = val; __syncthreads();
    for (int s = 64; s > 0; s >>= 1) { if (j < s) sh[j] += sh[j + s]; __syncthreads(); }
    float mean = sh[0] / (float)HID_; __syncthreads();
    float dv = val - mean;
    sh[j] = dv * dv; __syncthreads();
    for (int s = 64; s > 0; s >>= 1) { if (j < s) sh[j] += sh[j + s]; __syncthreads(); }
    float var = sh[0] / (float)HID_;
    out[node * ldo + j] = dv * rsqrtf(var + 1e-5f) * g[j] + b[j];
}

// ---------------- misc small kernels ----------------
__global__ void fill_f32_kernel(float* p, float v, int n) {
    int i = blockIdx.x * blockDim.x + threadIdx.x;
    if (i < n) p[i] = v;
}
__global__ void fill_u32_kernel(unsigned* p, unsigned v, int n) {
    int i = blockIdx.x * blockDim.x + threadIdx.x;
    if (i < n) p[i] = v;
}
__global__ void copy_strided_kernel(const float* __restrict__ src, int lds,
                                    float* __restrict__ dst, int ldd, int n, int w) {
    int idx = blockIdx.x * blockDim.x + threadIdx.x;
    if (idx >= n * w) return;
    int node = idx / w, j = idx - node * w;
    dst[node * ldd + j] = src[node * lds + j];
}
__global__ void softmax_rows_kernel(const float* __restrict__ in, int ldin,
                                    float* __restrict__ out, int n, int w) {
    int node = blockIdx.x * blockDim.x + threadIdx.x;
    if (node >= n) return;
    float m = -INFINITY;
    for (int j = 0; j < w; ++j) m = fmaxf(m, in[node * ldin + j]);
    float s = 0.0f, e[CLS_];
    for (int j = 0; j < w; ++j) { e[j] = __expf(in[node * ldin + j] - m); s += e[j]; }
    float inv = 1.0f / s;
    for (int j = 0; j < w; ++j) out[node * w + j] = e[j] * inv;
}

// =====================================================================
// host orchestration
// =====================================================================
struct GemmCtx { half_t* bpack; hipStream_t st; };

// pack B (with optional zero-padding) then run the WMMA GEMM
static inline void launch_gemm(const GemmCtx& cx, const float* A, int lda,
                               const float* Bsrc, int ldb, int Ksrc, int Ncsrc,
                               const float* bias, float* C, int ldc,
                               int M, int K, int Nc, int flags) {
    int packElems = K * Nc;
    pack_b_kernel<<<(packElems + 255) / 256, 256, 0, cx.st>>>(
        Bsrc, cx.bpack, K, Nc, Ksrc, Ncsrc, ldb);
    dim3 g(M / 16, Nc / 64);
    gemm_wmma_kernel<<<g, dim3(32), 0, cx.st>>>(A, lda, cx.bpack, bias, C, ldc,
                                                K, Nc, flags);
}

extern "C" void kernel_launch(void* const* d_in, const int* in_sizes, int n_in,
                              void* d_out, int out_size, void* d_ws, size_t ws_size,
                              hipStream_t stream) {
    (void)in_sizes; (void)n_in; (void)out_size; (void)ws_size;
    // ---- inputs (setup_inputs order) ----
    const float*     x       = (const float*)d_in[0];
    const long long* eidx    = (const long long*)d_in[1];  // [T,2,E] int64
    const float*     origA   = (const float*)d_in[2];
    const float*     lap     = (const float*)d_in[3];
    const float*     deg     = (const float*)d_in[4];
    const float*     W_trans = (const float*)d_in[5];
    const float*     b_trans = (const float*)d_in[6];
    const float*     W1      = (const float*)d_in[7];
    const float*     as1     = (const float*)d_in[8];
    const float*     ad1     = (const float*)d_in[9];
    const float*     b1      = (const float*)d_in[10];
    const float*     W2      = (const float*)d_in[11];
    const float*     as2     = (const float*)d_in[12];
    const float*     ad2     = (const float*)d_in[13];
    const float*     b2      = (const float*)d_in[14];
    const float*     peQ     = (const float*)d_in[15];
    const float*     peK     = (const float*)d_in[16];
    const float*     Wq      = (const float*)d_in[17];
    const float*     Wk      = (const float*)d_in[18];
    const float*     Wv      = (const float*)d_in[19];
    const float*     Wo      = (const float*)d_in[20];
    const float*     Wpq     = (const float*)d_in[21];
    const float*     Wpk     = (const float*)d_in[22];
    const float*     Wf1     = (const float*)d_in[23];
    const float*     bf1     = (const float*)d_in[24];
    const float*     Wf2     = (const float*)d_in[25];
    const float*     bf2     = (const float*)d_in[26];
    const float*     ln1g    = (const float*)d_in[27];
    const float*     ln1b    = (const float*)d_in[28];
    const float*     ln2g    = (const float*)d_in[29];
    const float*     ln2b    = (const float*)d_in[30];
    const float*     Wcat    = (const float*)d_in[31];
    const float*     bcat    = (const float*)d_in[32];
    const float*     Wm1     = (const float*)d_in[33];
    const float*     bm1     = (const float*)d_in[34];
    const float*     Wm2     = (const float*)d_in[35];
    const float*     alpha   = (const float*)d_in[36];

    // ---- output layout: (loss[1], logits[N,8], x_gt[N,128], pe_Q[N,112], pe_K[N,112]) ----
    float* out_loss   = (float*)d_out;
    float* out_logits = out_loss + 1;
    float* out_xgt    = out_logits + (size_t)N_ * CLS_;
    float* out_pq     = out_xgt + (size_t)N_ * HID_;
    float* out_pk     = out_pq + (size_t)N_ * PED_;

    // ---- workspace carve-out (all sizes multiples of 16 floats) ----
    float* w = (float*)d_ws;
    size_t off = 0;
    auto alloc = [&](size_t nfl) { float* p = w + off; off += nfl; return p; };
    float*    xp    = alloc((size_t)N_ * HID_);
    float*    xw    = alloc((size_t)N_ * H4C_);
    float*    hacc  = alloc((size_t)N_ * H4C_);
    float*    ssb   = alloc((size_t)N_ * H_);
    float*    sdb   = alloc((size_t)N_ * H_);
    unsigned* emaxb = (unsigned*)alloc((size_t)N_ * H_);
    float*    esumb = alloc((size_t)N_ * H_);
    float*    exb   = alloc((size_t)E_ * H_);
    float*    xgb   = alloc((size_t)N_ * HID_);
    float*    pqc   = alloc((size_t)N_ * PEDP_);    // zero-padded 112->128
    float*    pkc   = alloc((size_t)N_ * PEDP_);
    float*    qb    = alloc((size_t)N_ * HID_);
    float*    kbuf  = alloc((size_t)N_ * HID_);
    float*    vb    = alloc((size_t)N_ * HID_);
    float*    oattn = alloc((size_t)N_ * HID_);
    float*    oproj = alloc((size_t)N_ * HID_);
    float*    h1b   = alloc((size_t)N_ * HID_);
    float*    f1b   = alloc((size_t)N_ * 2 * HID_);
    float*    saveb = alloc((size_t)N_ * 3 * HID_); // [save0|save1|save2], ld=384
    float*    xgt   = alloc((size_t)N_ * HID_);
    float*    m1b   = alloc((size_t)N_ * 2 * HID_);
    float*    lgt   = alloc((size_t)N_ * CLSP_);    // padded logits
    half_t*   bpack = (half_t*)alloc((size_t)32768); // 65536 halfs (max K*Nc)
    float*    lacc  = alloc(1);

    const GemmCtx cx{bpack, stream};
    const int TPB = 256;
    auto blocks = [&](int n) { return (n + TPB - 1) / TPB; };

    // ---- 1. xp = relu(x @ W_trans + b_trans) ----
    launch_gemm(cx, x, IN_, W_trans, HID_, IN_, HID_, b_trans,
                xp, HID_, N_, IN_, HID_, GEMM_RELU);

    // ---- 2. hetero GAT layer 1 (heads=4, out per type [N,512], sum over types) ----
    fill_f32_kernel<<<blocks(N_ * H4C_), TPB, 0, stream>>>(hacc, 0.0f, N_ * H4C_);
    for (int t = 0; t < T_; ++t) {
        const long long* srcp = eidx + (size_t)t * 2 * E_;
        const long long* dstp = srcp + E_;
        launch_gemm(cx, xp, HID_, W1 + (size_t)t * HID_ * H4C_, H4C_, HID_, H4C_,
                    nullptr, xw, H4C_, N_, HID_, H4C_, 0);
        gat_scores_kernel<<<blocks(N_ * H_), TPB, 0, stream>>>(
            xw, as1 + (size_t)t * H_ * HID_, ad1 + (size_t)t * H_ * HID_,
            ssb, sdb, N_, H_, HID_);
        fill_u32_kernel<<<blocks(N_ * H_), TPB, 0, stream>>>(emaxb, ORD_NEG_INF, N_ * H_);
        fill_f32_kernel<<<blocks(N_ * H_), TPB, 0, stream>>>(esumb, 0.0f, N_ * H_);
        gat_edge_logit_max_kernel<<<blocks(E_ * H_), TPB, 0, stream>>>(
            srcp, dstp, ssb, sdb, exb, emaxb, E_, H_);
        gat_edge_exp_sum_kernel<<<blocks(E_ * H_), TPB, 0, stream>>>(
            dstp, exb, emaxb, esumb, E_, H_);
        gat_edge_scatter_kernel<<<E_, 256, 0, stream>>>(
            srcp, dstp, xw, exb, esumb, hacc, H_, HID_);
    }
    bias3_act_kernel<<<blocks(N_ * H4C_), TPB, 0, stream>>>(
        hacc, b1, b1 + H4C_, b1 + 2 * H4C_, N_, H4C_, 1);   // + relu

    // ---- 3. hetero GAT layer 2 (heads=1, [N,128], sum over types) ----
    fill_f32_kernel<<<blocks(N_ * HID_), TPB, 0, stream>>>(xgb, 0.0f, N_ * HID_);
    for (int t = 0; t < T_; ++t) {
        const long long* srcp = eidx + (size_t)t * 2 * E_;
        const long long* dstp = srcp + E_;
        launch_gemm(cx, hacc, H4C_, W2 + (size_t)t * H4C_ * HID_, HID_, H4C_, HID_,
                    nullptr, xw, HID_, N_, H4C_, HID_, 0);
        gat_scores_kernel<<<blocks(N_), TPB, 0, stream>>>(
            xw, as2 + (size_t)t * HID_, ad2 + (size_t)t * HID_, ssb, sdb, N_, 1, HID_);
        fill_u32_kernel<<<blocks(N_), TPB, 0, stream>>>(emaxb, ORD_NEG_INF, N_);
        fill_f32_kernel<<<blocks(N_), TPB, 0, stream>>>(esumb, 0.0f, N_);
        gat_edge_logit_max_kernel<<<blocks(E_), TPB, 0, stream>>>(
            srcp, dstp, ssb, sdb, exb, emaxb, E_, 1);
        gat_edge_exp_sum_kernel<<<blocks(E_), TPB, 0, stream>>>(
            dstp, exb, emaxb, esumb, E_, 1);
        gat_edge_scatter_kernel<<<E_, 128, 0, stream>>>(
            srcp, dstp, xw, exb, esumb, xgb, 1, HID_);
    }
    bias3_act_kernel<<<blocks(N_ * HID_), TPB, 0, stream>>>(
        xgb, b2, b2 + HID_, b2 + 2 * HID_, N_, HID_, 0);    // no relu

    // ---- 4. PE concat (zero-padded ws + exact d_out) and reconstruction loss ----
    fill_f32_kernel<<<blocks(N_ * PEDP_), TPB, 0, stream>>>(pqc, 0.0f, N_ * PEDP_);
    fill_f32_kernel<<<blocks(N_ * PEDP_), TPB, 0, stream>>>(pkc, 0.0f, N_ * PEDP_);
    pe_concat_kernel<<<blocks(N_ * PED_), TPB, 0, stream>>>(
        peQ, peK, lap, pqc, pkc, PEDP_, out_pq, out_pk, N_);
    fill_f32_kernel<<<1, 1, 0, stream>>>(lacc, 0.0f, 1);
    {
        int total = T_ * N_ * N_;
        pe_loss_kernel<<<blocks(total), TPB, 0, stream>>>(peQ, peK, origA, lacc, N_, total);
        loss_finalize_kernel<<<1, 32, 0, stream>>>(
            lacc, alpha, out_loss, 1.0f / ((float)N_ * (float)N_ * (float)T_));
    }

    // ---- 5. graph transformer (L=2) ----
    copy_strided_kernel<<<blocks(N_ * HID_), TPB, 0, stream>>>(
        xgb, HID_, saveb, 3 * HID_, N_, HID_);              // save0
    const float scale = 1.0f / sqrtf((float)DH_);
    for (int i = 0; i < L_; ++i) {
        const float* hh = saveb + (size_t)i * HID_;         // lda = 384
        launch_gemm(cx, hh, 3 * HID_, Wq + (size_t)i * HID_ * HID_, HID_, HID_, HID_,
                    nullptr, qb, HID_, N_, HID_, HID_, 0);
        launch_gemm(cx, pqc, PEDP_, Wpq + (size_t)i * PED_ * HID_, HID_, PED_, HID_,
                    nullptr, qb, HID_, N_, PEDP_, HID_, GEMM_ACC);   // K-padded
        launch_gemm(cx, hh, 3 * HID_, Wk + (size_t)i * HID_ * HID_, HID_, HID_, HID_,
                    nullptr, kbuf, HID_, N_, HID_, HID_, 0);
        launch_gemm(cx, pkc, PEDP_, Wpk + (size_t)i * PED_ * HID_, HID_, PED_, HID_,
                    nullptr, kbuf, HID_, N_, PEDP_, HID_, GEMM_ACC); // K-padded
        launch_gemm(cx, hh, 3 * HID_, Wv + (size_t)i * HID_ * HID_, HID_, HID_, HID_,
                    nullptr, vb, HID_, N_, HID_, HID_, 0);
        attn_flash_kernel<<<dim3(N_ / 16, H_), 32, 0, stream>>>(
            qb, kbuf, vb, oattn, N_, scale);
        launch_gemm(cx, oattn, HID_, Wo + (size_t)i * HID_ * HID_, HID_, HID_, HID_,
                    nullptr, oproj, HID_, N_, HID_, HID_, 0);
        resid_ln_kernel<<<N_, HID_, 0, stream>>>(
            hh, 3 * HID_, oproj, deg, ln1g + (size_t)i * HID_, ln1b + (size_t)i * HID_,
            h1b, HID_);
        launch_gemm(cx, h1b, HID_, Wf1 + (size_t)i * HID_ * 2 * HID_, 2 * HID_,
                    HID_, 2 * HID_, bf1 + (size_t)i * 2 * HID_,
                    f1b, 2 * HID_, N_, HID_, 2 * HID_, GEMM_RELU);
        launch_gemm(cx, f1b, 2 * HID_, Wf2 + (size_t)i * 2 * HID_ * HID_, HID_,
                    2 * HID_, HID_, bf2 + (size_t)i * HID_,
                    oproj, HID_, N_, 2 * HID_, HID_, 0);
        resid_ln_kernel<<<N_, HID_, 0, stream>>>(
            h1b, HID_, oproj, nullptr, ln2g + (size_t)i * HID_, ln2b + (size_t)i * HID_,
            saveb + (size_t)(i + 1) * HID_, 3 * HID_);       // save_{i+1}
    }

    // ---- 6. readout ----
    launch_gemm(cx, saveb, 3 * HID_, Wcat, HID_, 3 * HID_, HID_, bcat,
                xgt, HID_, N_, 3 * HID_, HID_, GEMM_RELU);
    copy_strided_kernel<<<blocks(N_ * HID_), TPB, 0, stream>>>(
        xgt, HID_, out_xgt, HID_, N_, HID_);
    launch_gemm(cx, xgt, HID_, Wm1, 2 * HID_, HID_, 2 * HID_, bm1,
                m1b, 2 * HID_, N_, HID_, 2 * HID_, GEMM_RELU);
    launch_gemm(cx, m1b, 2 * HID_, Wm2, CLS_, 2 * HID_, CLS_, nullptr,
                lgt, CLSP_, N_, 2 * HID_, CLSP_, 0);         // Nc-padded 8->64
    softmax_rows_kernel<<<blocks(N_), TPB, 0, stream>>>(lgt, CLSP_, out_logits, N_, CLS_);
}
